// BinaryTreeComposer_32873679684053
// MI455X (gfx1250) — matrix operand order repaired and checked by reading, not compiled, not run on hardware
//
#include <hip/hip_runtime.h>
#include <stdint.h>

#define B_SZ   16384
#define IN_DIM 1024
#define MEM    1024
#define NG     5

#define BM 128          // batch rows per workgroup
#define BN 64           // output cols per workgroup
#define BK 32           // k-step (== WMMA K)

typedef __attribute__((ext_vector_type(16))) __bf16 v16bf;
typedef __attribute__((ext_vector_type(8)))  __bf16 v8bf;
typedef __attribute__((ext_vector_type(8)))  float  v8f;

#define A_TILE_BYTES (BM * BK * 2)                        // 8192  (row = 64B = 4 chunks)
#define B_TILE_BYTES (BK * BN * 2)                        // 4096  (row = 128B = 8 chunks)
#define BUF_BYTES    (A_TILE_BYTES + NG * B_TILE_BYTES)   // 28672
#define LDS_BYTES    (2 * BUF_BYTES)                      // 57344 (< 64KB)

// ---------------- fp32 -> bf16 conversion pass ----------------
static __device__ __forceinline__ unsigned int f2bf(float f) {
  union { float f; unsigned int u; } v; v.f = f;
  unsigned int r = v.u + 0x7FFFu + ((v.u >> 16) & 1u);   // round-to-nearest-even
  return r >> 16;
}

__global__ void cvt_f32_to_bf16_x8(const float* __restrict__ s,
                                   uint4* __restrict__ d, int n8) {
  int i = blockIdx.x * blockDim.x + threadIdx.x;
  if (i >= n8) return;
  const float4* s4 = (const float4*)s;
  float4 a = s4[2 * i], b = s4[2 * i + 1];
  uint4 o;
  o.x = f2bf(a.x) | (f2bf(a.y) << 16);
  o.y = f2bf(a.z) | (f2bf(a.w) << 16);
  o.z = f2bf(b.x) | (f2bf(b.y) << 16);
  o.w = f2bf(b.z) | (f2bf(b.w) << 16);
  d[i] = o;
}

// ---------------- CDNA5 async copy global -> LDS ----------------
static __device__ __forceinline__ void async_g2l_b128(unsigned int ldsOff, const void* gaddr) {
  // GLOBAL_LOAD_ASYNC_TO_LDS_B128: vdst = LDS byte address, vaddr = 64-bit global addr
  asm volatile("global_load_async_to_lds_b128 %0, %1, off"
               :: "v"(ldsOff), "v"(gaddr) : "memory");
}
static __device__ __forceinline__ void wait_async0() {
  asm volatile("s_wait_asynccnt 0" ::: "memory");
}

static __device__ __forceinline__ v8f bf16mma(v16bf a, v16bf b, v8f c) {
  return __builtin_amdgcn_wmma_f32_16x16x32_bf16(false, a, false, b, (short)0, c, false, false);
}

static __device__ __forceinline__ v16bf cat8(v8bf lo, v8bf hi) {
  return __builtin_shufflevector(lo, hi, 0, 1, 2, 3, 4, 5, 6, 7,
                                 8, 9, 10, 11, 12, 13, 14, 15);
}

static __device__ __forceinline__ float sigmoidf_(float x) {
  return 1.0f / (1.0f + __expf(-x));
}
static __device__ __forceinline__ float tanhf_(float x) {
  return 2.0f / (1.0f + __expf(-2.0f * x)) - 1.0f;
}

// One K=1024 phase: double-buffered async-staged pipeline, NB weight tiles.
// LDS layout per buffer: [A 128x32 bf16, 64B rows, chunk c at (c ^ (row&3))]
//                        [NB x B 32x64 bf16, 128B rows, chunk c at (c ^ (row&7))]
template <int NB>
static __device__ __forceinline__ void
phase_loop(const unsigned short* __restrict__ Asrc,
           const unsigned short* __restrict__ Wsrc,
           const char* ldsBase, unsigned int ldsAddr,
           int blockM, int blockN, int tid, int lane, int wm, int wn,
           v8f (&acc)[NB][4]) {
  const int aRow0 = tid >> 2, aC = tid & 3;   // A staging: 2 chunks/thread
  const int bRow  = tid >> 3, bC = tid & 7;   // B staging: 1 chunk/thread/tile

  auto issue = [&](int buf, int kk) {
    const unsigned int bufA = ldsAddr + buf * BUF_BYTES;
#pragma unroll
    for (int it = 0; it < 2; ++it) {
      const int row = aRow0 + it * 64;
      const unsigned int l = bufA + row * 64 + (((aC ^ row) & 3) << 4);
      async_g2l_b128(l, Asrc + (size_t)(blockM + row) * MEM + kk + aC * 8);
    }
    const unsigned int bufB = bufA + A_TILE_BYTES + bRow * 128 + (((bC ^ bRow) & 7) << 4);
#pragma unroll
    for (int t = 0; t < NB; ++t) {
      async_g2l_b128(bufB + t * B_TILE_BYTES,
                     Wsrc + ((size_t)t << 20) + (size_t)(kk + bRow) * MEM + blockN + bC * 8);
    }
  };

  auto compute = [&](int buf) {
    const char* p = ldsBase + buf * BUF_BYTES;
    // A fragments: lane = M row; lanes 0-15 hold K{0-7,16-23}, lanes 16-31 K{8-15,24-31}
    v16bf a[2];
#pragma unroll
    for (int mi = 0; mi < 2; ++mi) {
      const int row = wm * 32 + mi * 16 + (lane & 15);
      const int c0 = (lane >> 4);        // chunk of first 16B (K0-7 / K8-15)
      const char* rp = p + row * 64;
      v8bf lo = *(const v8bf*)(rp + (((c0    ) ^ (row & 3)) << 4));
      v8bf hi = *(const v8bf*)(rp + (((c0 + 2) ^ (row & 3)) << 4));
      a[mi] = cat8(lo, hi);
    }
    // B fragments: lane = N col (lane%16); lanes 0-15 hold K{0-7,16-23}, lanes 16-31 K{8-15,24-31}
#pragma unroll
    for (int t = 0; t < NB; ++t) {
      const char* bp = p + A_TILE_BYTES + t * B_TILE_BYTES;
      v16bf b[2];
#pragma unroll
      for (int ni = 0; ni < 2; ++ni) {
        const int n = wn * 32 + ni * 16 + (lane & 15);
#pragma unroll
        for (int s = 0; s < 16; ++s) {
          const int k = (s < 8) ? (8 * (lane >> 4) + s) : (16 + 8 * (lane >> 4) + (s - 8));
          b[ni][s] = *(const __bf16*)(bp + k * 128 + ((((n >> 3) ^ k) & 7) << 4) + (n & 7) * 2);
        }
      }
      acc[t][0] = bf16mma(a[0], b[0], acc[t][0]);
      acc[t][1] = bf16mma(a[0], b[1], acc[t][1]);
      acc[t][2] = bf16mma(a[1], b[0], acc[t][2]);
      acc[t][3] = bf16mma(a[1], b[1], acc[t][3]);
    }
  };

  issue(0, 0);
  for (int kt = 0; kt < 32; ++kt) {
    wait_async0();        // my async stores into buffer (kt&1) are done
    __syncthreads();      // everyone's are done; also fences last compute on (kt+1)&1
    if (kt + 1 < 32) issue((kt + 1) & 1, (kt + 1) * BK);
    compute(kt & 1);
  }
}

// Fused: pre_g = input@Wi + bi + lh@Wlh_g + blh_g + rh@Wrh_g + brh_g  (g=0..4)
// c = sig(p0)*tanh(p4) + sig(p1)*lc + sig(p2)*rc ;  h = sig(p3)*tanh(c)
__global__ __launch_bounds__(256, 1)
void treelstm_fused(const unsigned short* __restrict__ Xb,
                    const unsigned short* __restrict__ LHb,
                    const unsigned short* __restrict__ RHb,
                    const unsigned short* __restrict__ Wib,
                    const unsigned short* __restrict__ Wlhb,
                    const unsigned short* __restrict__ Wrhb,
                    const float* __restrict__ bi,
                    const float* __restrict__ blh,
                    const float* __restrict__ brh,
                    const float* __restrict__ lc,
                    const float* __restrict__ rc,
                    float* __restrict__ outC,
                    float* __restrict__ outH) {
  __shared__ __attribute__((aligned(16))) char lds[LDS_BYTES];

  const int tid  = threadIdx.x;
  const int lane = tid & 31;
  const int wid  = tid >> 5;
  const int wm   = wid & 3;                  // 4 waves along M
  const int wn   = wid >> 2;                 // 2 waves along N
  const int blockM = blockIdx.x * BM;
  const int blockN = blockIdx.y * BN;

  const char* ldsBase = lds;
  const unsigned int ldsAddr = (unsigned int)(size_t)(void*)lds;   // LDS byte offset

  v8f accX[1][4];
  v8f accG[NG][4];
  const v8f vzero = {0.f, 0.f, 0.f, 0.f, 0.f, 0.f, 0.f, 0.f};
#pragma unroll
  for (int t = 0; t < 4; ++t) accX[0][t] = vzero;
#pragma unroll
  for (int g = 0; g < NG; ++g)
#pragma unroll
    for (int t = 0; t < 4; ++t) accG[g][t] = vzero;

  phase_loop<1>(Xb,  Wib,  ldsBase, ldsAddr, blockM, blockN, tid, lane, wm, wn, accX);
  phase_loop<NG>(LHb, Wlhb, ldsBase, ldsAddr, blockM, blockN, tid, lane, wm, wn, accG);
  phase_loop<NG>(RHb, Wrhb, ldsBase, ldsAddr, blockM, blockN, tid, lane, wm, wn, accG);

  // --- fused epilogue ---
  // C/D layout: VGPR r -> row M = r (lanes 0-15) or 8+r (lanes 16-31); col N = lane%16
  const int colL = lane & 15;
  const int rAdd = (lane >> 4) * 8;
#pragma unroll
  for (int mi = 0; mi < 2; ++mi) {
#pragma unroll
    for (int ni = 0; ni < 2; ++ni) {
      const int accIdx = mi * 2 + ni;
      const int gn = blockN + wn * 32 + ni * 16 + colL;
      const float bival = bi[gn];
      float bsum[NG];
#pragma unroll
      for (int g = 0; g < NG; ++g)
        bsum[g] = bival + blh[g * MEM + gn] + brh[g * MEM + gn];
#pragma unroll
      for (int r = 0; r < 8; ++r) {
        const int gm = blockM + wm * 32 + mi * 16 + rAdd + r;
        const size_t off = (size_t)gm * MEM + gn;
        const float xiv = accX[0][accIdx][r];
        const float p0 = accG[0][accIdx][r] + xiv + bsum[0];
        const float p1 = accG[1][accIdx][r] + xiv + bsum[1];
        const float p2 = accG[2][accIdx][r] + xiv + bsum[2];
        const float p3 = accG[3][accIdx][r] + xiv + bsum[3];
        const float p4 = accG[4][accIdx][r] + xiv + bsum[4];
        const float iG  = sigmoidf_(p0);
        const float lfG = sigmoidf_(p1);
        const float rfG = sigmoidf_(p2);
        const float oG  = sigmoidf_(p3);
        const float uG  = tanhf_(p4);
        const float cNew = iG * uG + lfG * lc[off] + rfG * rc[off];
        const float hNew = oG * tanhf_(cNew);
        outC[off] = cNew;
        outH[off] = hNew;
      }
    }
  }
}

extern "C" void kernel_launch(void* const* d_in, const int* in_sizes, int n_in,
                              void* d_out, int out_size, void* d_ws, size_t ws_size,
                              hipStream_t stream) {
  const float* input = (const float*)d_in[0];
  const float* lc    = (const float*)d_in[1];
  const float* lh    = (const float*)d_in[2];
  const float* rc    = (const float*)d_in[3];
  const float* rh    = (const float*)d_in[4];
  const float* Wi    = (const float*)d_in[5];
  const float* bi    = (const float*)d_in[6];
  const float* Wlh   = (const float*)d_in[7];
  const float* blh   = (const float*)d_in[8];
  const float* Wrh   = (const float*)d_in[9];
  const float* brh   = (const float*)d_in[10];

  // workspace: bf16 copies of all GEMM operands (~120MB; re-reads then hit 192MB L2)
  char* ws = (char*)d_ws;
  unsigned short* Xb   = (unsigned short*)ws; ws += (size_t)B_SZ * IN_DIM * 2;
  unsigned short* LHb  = (unsigned short*)ws; ws += (size_t)B_SZ * MEM * 2;
  unsigned short* RHb  = (unsigned short*)ws; ws += (size_t)B_SZ * MEM * 2;
  unsigned short* Wib  = (unsigned short*)ws; ws += (size_t)IN_DIM * MEM * 2;
  unsigned short* Wlhb = (unsigned short*)ws; ws += (size_t)NG * MEM * MEM * 2;
  unsigned short* Wrhb = (unsigned short*)ws; ws += (size_t)NG * MEM * MEM * 2;

  auto cvt = [&](const float* s, unsigned short* d, size_t n) {
    int n8 = (int)(n / 8);
    cvt_f32_to_bf16_x8<<<(n8 + 255) / 256, 256, 0, stream>>>(s, (uint4*)d, n8);
  };
  cvt(input, Xb,   (size_t)B_SZ * IN_DIM);
  cvt(lh,    LHb,  (size_t)B_SZ * MEM);
  cvt(rh,    RHb,  (size_t)B_SZ * MEM);
  cvt(Wi,    Wib,  (size_t)IN_DIM * MEM);
  cvt(Wlh,   Wlhb, (size_t)NG * MEM * MEM);
  cvt(Wrh,   Wrhb, (size_t)NG * MEM * MEM);

  float* outC = (float*)d_out;
  float* outH = outC + (size_t)B_SZ * MEM;

  dim3 grid(B_SZ / BM, MEM / BN);   // 128 x 16
  treelstm_fused<<<grid, 256, 0, stream>>>(Xb, LHb, RHb, Wib, Wlhb, Wrhb,
                                           bi, blh, brh, lc, rc, outC, outH);
}
